// DDGraphUNet_86766929314321
// MI455X (gfx1250) — matrix operand order repaired and checked by reading, not compiled, not run on hardware
//
#include <hip/hip_runtime.h>
#include <math.h>

// ---------------------------------------------------------------------------
// CDNA5 (gfx1250) DDGraphUNet forward. All dense contractions run through
// v_wmma_f32_16x16x32_f16 (f16 inputs, f32 accumulate).
//  - Weights pre-packed (once per launch) into per-lane WMMA B-fragment order.
//  - GEMM A-operands padded to K multiple of 32 -> unconditional vector loads,
//    staged to LDS in A-fragment order (32B contiguous per lane).
// ---------------------------------------------------------------------------

typedef __attribute__((ext_vector_type(16))) _Float16 v16h;
typedef __attribute__((ext_vector_type(4)))  _Float16 v4h;
typedef __attribute__((ext_vector_type(8)))  float    v8f;

#define ACT_NONE 0
#define ACT_GELU 1
#define ACT_RELU 2

__device__ __forceinline__ float gelu_f(float x) {
  return 0.5f * x * (1.0f + erff(x * 0.70710678118654752440f));
}

__device__ __forceinline__ v8f wmma16x16x32(v16h a, v16h b, v8f c) {
  return __builtin_amdgcn_wmma_f32_16x16x32_f16(false, a, false, b, (short)0, c,
                                                false, false);
}

// ---------------------------------------------------------------------------
// Pack W[K,M] (f32, row-major) -> B-fragment layout f16:
//   Wf[((ct*Kt + kc)*32 + lane)*16 + 2v+p] = W[kc*32 + 16*(lane>>4) + 2v + p][ct*16 + (lane&15)]
// Zero padded beyond K / M.
// ---------------------------------------------------------------------------
__global__ void k_pack_w(const float* __restrict__ W, int K, int M,
                         int Kt, int Mt, _Float16* __restrict__ Wf)
{
  int gid = blockIdx.x * blockDim.x + threadIdx.x;
  int total = Mt * Kt * 32;
  if (gid >= total) return;
  int lane = gid & 31;
  int kc   = (gid >> 5) % Kt;
  int ct   = gid / (32 * Kt);
  int n = lane & 15, g = lane >> 4;
  int col = ct * 16 + n;
  _Float16* dst = Wf + (size_t)gid * 16;
  #pragma unroll
  for (int v = 0; v < 8; ++v) {
    int k = kc * 32 + 16 * g + 2 * v;
    float f0 = (k     < K && col < M) ? W[(size_t)k * M + col]       : 0.0f;
    float f1 = (k + 1 < K && col < M) ? W[(size_t)(k + 1) * M + col] : 0.0f;
    dst[2 * v]     = (_Float16)f0;
    dst[2 * v + 1] = (_Float16)f1;
  }
}

// ---------------------------------------------------------------------------
// GEMM: C[N,M] = act(A[N,lda] @ W + bias) (+= old if accum).
// Requirements: N % 128 == 0, lda = Kt*32 (zero padded), Wf pre-packed.
// Block = 256 threads (8 waves); block tile = 128 rows x 16 cols; wave w owns
// rows [w*16, w*16+16). A staged into LDS in fragment order.
// ---------------------------------------------------------------------------
__global__ __launch_bounds__(256) void k_gemm(
    const float* __restrict__ A,
    const _Float16* __restrict__ Wf, int Kt,
    const float* __restrict__ bias,
    float* __restrict__ C, int M, int act, int accum)
{
  __shared__ _Float16 As[8][32][16];   // [rowTile][lane][16 halves]
  const int tid  = threadIdx.x;
  const int wave = tid >> 5;
  const int lane = tid & 31;
  const int rowBase = blockIdx.x * 128;
  const int lda = Kt * 32;
  const _Float16* wcol = Wf + (size_t)blockIdx.y * Kt * 32 * 16;

  v8f acc = {};
  for (int kc = 0; kc < Kt; ++kc) {
    // stage A chunk (128 x 32) into fragment layout: 4 float4 per thread
    #pragma unroll
    for (int i = 0; i < 4; ++i) {
      int e = tid * 4 + i;              // 1024 float4's
      int r = e >> 3;                   // row 0..127
      int k4 = (e & 7) * 4;             // 0,4,...,28
      const float4 f = *(const float4*)(A + (size_t)(rowBase + r) * lda + kc * 32 + k4);
      int g  = k4 >> 4;
      int va = (k4 & 15) >> 1;     // even
      v4h h = { (_Float16)f.x, (_Float16)f.y, (_Float16)f.z, (_Float16)f.w };
      *(v4h*)&As[r >> 4][g * 16 + (r & 15)][2 * va] = h;
    }
    __syncthreads();
    v16h a = *(const v16h*)&As[wave][lane][0];
    v16h b = *(const v16h*)(wcol + ((size_t)kc * 32 + lane) * 16);
    acc = wmma16x16x32(a, b, acc);
    __syncthreads();
  }

  const int n = lane & 15, grp = lane >> 4;
  const int col = blockIdx.y * 16 + n;
  if (col < M) {
    float bv = bias ? bias[col] : 0.0f;
    #pragma unroll
    for (int r = 0; r < 8; ++r) {
      int row = rowBase + wave * 16 + r + 8 * grp;
      float v = acc[r] + bv;
      if (act == ACT_GELU) v = gelu_f(v);
      else if (act == ACT_RELU) v = fmaxf(v, 0.0f);
      size_t o = (size_t)row * M + col;
      if (accum) v += C[o];
      C[o] = v;
    }
  }
}

// ---------------------------------------------------------------------------
// Cluster aggregation m=256, d=128 (down0/op1):
//   agg[i,:] = (1/256) * sum_j ||p_i - p_j|| * hs[j,:], j in cluster(i).
// One block per cluster; hs tile swizzled in LDS to B-fragment order so each
// B-fragment is one contiguous 32B LDS read. Distance A-fragments in regs.
// ---------------------------------------------------------------------------
__global__ __launch_bounds__(256) void k_agg256(
    const float* __restrict__ hs, const float* __restrict__ pos,
    const int* __restrict__ members, float* __restrict__ agg)
{
  __shared__ _Float16 HsF[64][32][16];   // [(ct*8 + kc)][lane][16]
  __shared__ float Px[256], Py[256];
  __shared__ int Idx[256];
  const int c = blockIdx.x;
  const int tid = threadIdx.x;
  const int node = members[c * 256 + tid];
  Idx[tid] = node;
  Px[tid] = pos[(size_t)node * 2];
  Py[tid] = pos[(size_t)node * 2 + 1];
  const float* hrow = hs + (size_t)node * 128;
  // scatter this node's row into fragment order (node index == k index j)
  {
    int kcj  = tid >> 5;
    int krem = tid & 31;
    int gj = krem >> 4, vj = (krem & 15) >> 1, pj = krem & 1;
    #pragma unroll 4
    for (int ch = 0; ch < 128; ++ch) {
      int ct = ch >> 4, n = ch & 15;
      HsF[(ct << 3) + kcj][gj * 16 + n][2 * vj + pj] = (_Float16)hrow[ch];
    }
  }
  __syncthreads();

  const int wave = tid >> 5, lane = tid & 31;
  const int mrow = lane & 15, grp = lane >> 4;

  for (int t = 0; t < 2; ++t) {
    const int rt = wave * 2 + t;
    const int irow = rt * 16 + mrow;
    const float pix = Px[irow], piy = Py[irow];
    v8f acc[8];
    #pragma unroll
    for (int q = 0; q < 8; ++q) acc[q] = (v8f){0.f,0.f,0.f,0.f,0.f,0.f,0.f,0.f};

    for (int kc = 0; kc < 8; ++kc) {
      v16h a;
      #pragma unroll
      for (int v = 0; v < 8; ++v) {
        int ka = kc * 32 + 16 * grp + 2 * v;
        float dx0 = pix - Px[ka],     dy0 = piy - Py[ka];
        float dx1 = pix - Px[ka + 1], dy1 = piy - Py[ka + 1];
        a[2 * v]     = (_Float16)sqrtf(dx0 * dx0 + dy0 * dy0);
        a[2 * v + 1] = (_Float16)sqrtf(dx1 * dx1 + dy1 * dy1);
      }
      #pragma unroll
      for (int ct = 0; ct < 8; ++ct) {
        v16h b = *(const v16h*)&HsF[(ct << 3) + kc][lane][0];
        acc[ct] = wmma16x16x32(a, b, acc[ct]);
      }
    }
    #pragma unroll
    for (int r = 0; r < 8; ++r) {
      int lrow = rt * 16 + r + 8 * grp;
      float* arow = agg + (size_t)Idx[lrow] * 128;
      #pragma unroll
      for (int ct = 0; ct < 8; ++ct)
        arow[ct * 16 + (lane & 15)] = acc[ct][r] * (1.0f / 256.0f);
    }
  }
}

// Small-m cluster aggregation (bn1 m=16, op0 m=4). One thread per (node,ch).
__global__ void k_agg_small(const float* __restrict__ hs,
                            const float* __restrict__ pos,
                            const int* __restrict__ members,
                            const int* __restrict__ cl,
                            float* __restrict__ agg, int Ntot, int m)
{
  int gid = blockIdx.x * blockDim.x + threadIdx.x;
  if (gid >= Ntot * 128) return;
  int node = gid / 128, ch = gid % 128;
  int c = cl[node];
  float px = pos[(size_t)node * 2], py = pos[(size_t)node * 2 + 1];
  float s = 0.0f;
  for (int j = 0; j < m; ++j) {
    int nj = members[(size_t)c * m + j];
    float dx = px - pos[(size_t)nj * 2], dy = py - pos[(size_t)nj * 2 + 1];
    s += sqrtf(dx * dx + dy * dy) * hs[(size_t)nj * 128 + ch];
  }
  agg[gid] = s / (float)m;
}

// ------------------------- clustering helpers ------------------------------
__global__ void k_compute_cl(const float* __restrict__ pos, int n, int npg,
                             int nx, int ny, int* __restrict__ cl)
{
  int i = blockIdx.x * blockDim.x + threadIdx.x;
  if (i >= n) return;
  int b = i / npg;
  int ix = (int)(pos[(size_t)i * 2] * (float)nx);
  int iy = (int)(pos[(size_t)i * 2 + 1] * (float)ny);
  ix = ix < 0 ? 0 : (ix > nx - 1 ? nx - 1 : ix);
  iy = iy < 0 ? 0 : (iy > ny - 1 ? ny - 1 : iy);
  cl[i] = b * (nx * ny) + iy * nx + ix;
}

__global__ void k_fill_int(int* __restrict__ p, int n, int v)
{
  int i = blockIdx.x * blockDim.x + threadIdx.x;
  if (i < n) p[i] = v;
}

__global__ void k_bin(const int* __restrict__ cl, int n, int m,
                      int* __restrict__ cnt, int* __restrict__ members)
{
  int i = blockIdx.x * blockDim.x + threadIdx.x;
  if (i >= n) return;
  int c = cl[i];
  int slot = atomicAdd(&cnt[c], 1);
  members[(size_t)c * m + slot] = i;
}

// Bitonic sort member lists (m power of two, <=256) -> deterministic order.
__global__ void k_sort_members(int* __restrict__ members, int m)
{
  __shared__ int s[256];
  int t = threadIdx.x;
  int* seg = members + (size_t)blockIdx.x * m;
  if (t < m) s[t] = seg[t];
  __syncthreads();
  for (int k = 2; k <= m; k <<= 1) {
    for (int j = k >> 1; j > 0; j >>= 1) {
      int ixj = t ^ j;
      if (t < m && ixj > t) {
        int a = s[t], b = s[ixj];
        bool asc = ((t & k) == 0);
        if ((a > b) == asc) { s[t] = b; s[ixj] = a; }
      }
      __syncthreads();
    }
  }
  if (t < m) seg[t] = s[t];
}

// Mean-pool per cluster (sequential per channel over sorted list).
__global__ void k_pool_mean(const float* __restrict__ x, int w,
                            const int* __restrict__ members, int m,
                            float* __restrict__ out)
{
  int c = blockIdx.x, t = threadIdx.x;
  if (t >= w) return;
  float s = 0.0f;
  for (int j = 0; j < m; ++j)
    s += x[(size_t)members[(size_t)c * m + j] * w + t];
  out[(size_t)c * w + t] = s / (float)m;
}

// ------------------------- normalization -----------------------------------
__global__ __launch_bounds__(256) void k_inorm(const float* __restrict__ in,
                                               float* __restrict__ out,
                                               int npg, int C, int act)
{
  __shared__ float ssum[256], ssq[256];
  int b = blockIdx.x / C, ch = blockIdx.x % C;
  const float* base = in + (size_t)b * npg * C + ch;
  float s = 0.0f, q = 0.0f;
  for (int i = threadIdx.x; i < npg; i += 256) {
    float v = base[(size_t)i * C];
    s += v; q += v * v;
  }
  ssum[threadIdx.x] = s; ssq[threadIdx.x] = q;
  __syncthreads();
  for (int o = 128; o > 0; o >>= 1) {
    if (threadIdx.x < o) {
      ssum[threadIdx.x] += ssum[threadIdx.x + o];
      ssq[threadIdx.x]  += ssq[threadIdx.x + o];
    }
    __syncthreads();
  }
  float mu = ssum[0] / (float)npg;
  float var = fmaxf(ssq[0] / (float)npg - mu * mu, 0.0f);
  float inv = rsqrtf(var + 1e-5f);
  float* ob = out + (size_t)b * npg * C + ch;
  for (int i = threadIdx.x; i < npg; i += 256) {
    float v = (base[(size_t)i * C] - mu) * inv;
    if (act == ACT_GELU) v = gelu_f(v);
    ob[(size_t)i * C] = v;
  }
}

__global__ __launch_bounds__(256) void k_colstats(const float* __restrict__ in,
                                                  int N, int C,
                                                  float* __restrict__ mu,
                                                  float* __restrict__ var)
{
  __shared__ float ssum[256], ssq[256];
  int ch = blockIdx.x;
  float s = 0.0f, q = 0.0f;
  for (int i = threadIdx.x; i < N; i += 256) {
    float v = in[(size_t)i * C + ch];
    s += v; q += v * v;
  }
  ssum[threadIdx.x] = s; ssq[threadIdx.x] = q;
  __syncthreads();
  for (int o = 128; o > 0; o >>= 1) {
    if (threadIdx.x < o) {
      ssum[threadIdx.x] += ssum[threadIdx.x + o];
      ssq[threadIdx.x]  += ssq[threadIdx.x + o];
    }
    __syncthreads();
  }
  if (threadIdx.x == 0) {
    float m = ssum[0] / (float)N;
    mu[ch] = m;
    var[ch] = fmaxf(ssq[0] / (float)N - m * m, 0.0f);
  }
}

__global__ void k_bn_relu(const float* __restrict__ in, float* __restrict__ out,
                          int N, int C, const float* __restrict__ mu,
                          const float* __restrict__ var,
                          const float* __restrict__ g,
                          const float* __restrict__ bt)
{
  int gid = blockIdx.x * blockDim.x + threadIdx.x;
  if (gid >= N * C) return;
  int ch = gid % C;
  float v = (in[gid] - mu[ch]) * rsqrtf(var[ch] + 1e-5f) * g[ch] + bt[ch];
  out[gid] = fmaxf(v, 0.0f);
}

// ------------------------- knn interpolate (k=3) ----------------------------
__global__ void k_knn3_find(const float* __restrict__ ps,
                            const float* __restrict__ pt, int ns, int nt,
                            int totT, int* __restrict__ idx3,
                            float* __restrict__ w3)
{
  int t = blockIdx.x * blockDim.x + threadIdx.x;
  if (t >= totT) return;
  int b = t / nt;
  float tx = pt[(size_t)t * 2], ty = pt[(size_t)t * 2 + 1];
  const float* p = ps + (size_t)b * ns * 2;
  float d0 = 3.4e38f, d1 = 3.4e38f, d2 = 3.4e38f;
  int i0 = 0, i1 = 0, i2 = 0;
  for (int j = 0; j < ns; ++j) {
    float dx = tx - p[2 * j], dy = ty - p[2 * j + 1];
    float d = dx * dx + dy * dy;
    if (d < d0)      { d2 = d1; i2 = i1; d1 = d0; i1 = i0; d0 = d; i0 = j; }
    else if (d < d1) { d2 = d1; i2 = i1; d1 = d; i1 = j; }
    else if (d < d2) { d2 = d; i2 = j; }
  }
  float w0 = 1.0f / fmaxf(d0, 1e-16f);
  float w1 = 1.0f / fmaxf(d1, 1e-16f);
  float w2 = 1.0f / fmaxf(d2, 1e-16f);
  float ws = w0 + w1 + w2;
  idx3[3 * t]     = b * ns + i0;
  idx3[3 * t + 1] = b * ns + i1;
  idx3[3 * t + 2] = b * ns + i2;
  w3[3 * t] = w0 / ws; w3[3 * t + 1] = w1 / ws; w3[3 * t + 2] = w2 / ws;
}

__global__ void k_knn3_apply(const float* __restrict__ xs,
                             const int* __restrict__ idx3,
                             const float* __restrict__ w3,
                             float* __restrict__ out, int totT, int C)
{
  int gid = blockIdx.x * blockDim.x + threadIdx.x;
  if (gid >= totT * C) return;
  int t = gid / C, ch = gid % C;
  out[gid] = w3[3 * t]     * xs[(size_t)idx3[3 * t]     * C + ch]
           + w3[3 * t + 1] * xs[(size_t)idx3[3 * t + 1] * C + ch]
           + w3[3 * t + 2] * xs[(size_t)idx3[3 * t + 2] * C + ch];
}

// ------------------------- elementwise --------------------------------------
// Concat [a | b | zero-pad] into row stride ldo.
__global__ void k_concat2(const float* __restrict__ a, int wa,
                          const float* __restrict__ b, int wb,
                          float* __restrict__ out, int N, int ldo)
{
  int gid = blockIdx.x * blockDim.x + threadIdx.x;
  if (gid >= N * ldo) return;
  int r = gid / ldo, c = gid % ldo;
  float v = 0.0f;
  if (c < wa) v = a[(size_t)r * wa + c];
  else if (c < wa + wb) v = b[(size_t)r * wb + (c - wa)];
  out[gid] = v;
}

// Copy src[N,K] -> dst[N,ldo] with zero pad.
__global__ void k_pad_rows(const float* __restrict__ src, int K, int ldo,
                           float* __restrict__ dst, int N)
{
  int gid = blockIdx.x * blockDim.x + threadIdx.x;
  if (gid >= N * ldo) return;
  int r = gid / ldo, c = gid % ldo;
  dst[gid] = (c < K) ? src[(size_t)r * K + c] : 0.0f;
}

__global__ void k_add(const float* __restrict__ a, const float* __restrict__ b,
                      float* __restrict__ out, int n)
{
  int gid = blockIdx.x * blockDim.x + threadIdx.x;
  if (gid < n) out[gid] = a[gid] + b[gid];
}

// ---------------------------------------------------------------------------
struct DDP { const float *We1,*We2,*Wrel,*Wroot,*Wskip,*be1,*be2,*brel; };
struct MLPP { const float *W1,*W2,*b1,*b2,*bt,*g; };
struct PW { const _Float16* w; int Kt; int Mt; };
struct DDPW { PW We1, We2, Wrel, Wroot, Wskip; };

extern "C" void kernel_launch(void* const* d_in, const int* in_sizes, int n_in,
                              void* d_out, int out_size, void* d_ws, size_t ws_size,
                              hipStream_t stream)
{
  const int Bg = 8, C = 64;
  const int N0 = in_sizes[0] / 3;      // 131072
  const int NPG0 = N0 / Bg;            // 16384
  const int N1 = Bg * 64;              // 512
  const int N2 = Bg * 16;              // 128
  const int CHUNK = NPG0;

  const float* x   = (const float*)d_in[0];
  const float* pos = (const float*)d_in[1];
  // d_in[2] = batch (unused; graphs contiguous, equal size)

  // params flattened in jax pytree order (dict keys sorted alphabetically)
  int pi = 3;
  auto nextp = [&]() { return (const float*)d_in[pi++]; };
  auto getddp = [&]() {
    DDP d; d.We1=nextp(); d.We2=nextp(); d.Wrel=nextp(); d.Wroot=nextp();
    d.Wskip=nextp(); d.be1=nextp(); d.be2=nextp(); d.brel=nextp(); return d;
  };
  auto getmlp = [&]() {
    MLPP m; m.W1=nextp(); m.W2=nextp(); m.b1=nextp(); m.b2=nextp();
    m.bt=nextp(); m.g=nextp(); return m;
  };
  DDP bn1   = getddp();
  DDP down0 = getddp();
  DDP down1 = getddp();
  DDP down2 = getddp();
  const float *liftW1=nextp(), *liftW2=nextp(), *liftb1=nextp(), *liftb2=nextp();
  DDP op0 = getddp();
  DDP op1 = getddp();
  const float *projW1=nextp(), *projW2=nextp(), *projb1=nextp(), *projb2=nextp();
  MLPP up0 = getmlp();
  MLPP up1 = getmlp();

  // ---------------- workspace layout (floats, then ints, then f16) ----------
  float* F = (float*)d_ws;
  size_t o = 0;
  float* x0   = F + o; o += (size_t)N0 * C;
  float* bufA = F + o; o += (size_t)N0 * 2 * C;
  float* bufB = F + o; o += (size_t)N0 * 2 * C;
  float* bufC = F + o; o += (size_t)N0 * 96;   // padded concat / x-pad
  float* bufD = F + o; o += (size_t)N0 * C;    // >= CHUNK*256
  float* bufE = F + o; o += (size_t)N0 * C;
  float* knnw = F + o; o += (size_t)N0 * 3;
  float* stats= F + o; o += 256;
  float* pos1 = F + o; o += (size_t)N1 * 2;
  float* pos2 = F + o; o += (size_t)N2 * 2;
  float* x1p  = F + o; o += (size_t)N1 * C;
  float* x1   = F + o; o += (size_t)N1 * C;
  float* sInp = F + o; o += (size_t)N1 * 96;
  float* sT   = F + o; o += (size_t)N1 * C;
  float* sHd  = F + o; o += (size_t)N1 * 2 * C;
  float* sOut = F + o; o += (size_t)N1 * C;
  float* sAgg = F + o; o += (size_t)N1 * 2 * C;
  float* x2p  = F + o; o += (size_t)N2 * C;
  float* x2   = F + o; o += (size_t)N2 * C;
  float* hb   = F + o; o += (size_t)N2 * C;
  float* xi0  = F + o; o += (size_t)N1 * C;
  float* cat0 = F + o; o += (size_t)N1 * 2 * C;
  float* xm0  = F + o; o += (size_t)N1 * C;
  float* opin0= F + o; o += (size_t)N1 * C;
  float* xo1  = F + o; o += (size_t)N1 * C;
  int* I = (int*)(F + o);
  size_t io = 0;
  int* cl0  = I + io; io += N0;
  int* mem0 = I + io; io += N0;
  int* cl1  = I + io; io += N1;
  int* mem1 = I + io; io += N1;
  int* cl2  = I + io; io += N2;
  int* mem2 = I + io; io += N2;
  int* cnt  = I + io; io += 512;
  int* knni = I + io; io += (size_t)N0 * 3;
  size_t haddr = ((size_t)(I + io) + 31) & ~(size_t)31;
  _Float16* hptr = (_Float16*)haddr;
  (void)ws_size; (void)n_in; (void)out_size;

  auto ew = [&](long long n) { return dim3((unsigned)((n + 255) / 256)); };

  // -------- pack all weights into WMMA B-fragment layout (f16) --------
  auto packw = [&](const float* W, int K, int M) -> PW {
    int Kt = (K + 31) / 32, Mt = (M + 15) / 16;
    size_t nh = (size_t)Kt * Mt * 512;
    _Float16* p = hptr; hptr += nh;
    int total = Mt * Kt * 32;
    k_pack_w<<<ew(total), 256, 0, stream>>>(W, K, M, Kt, Mt, p);
    return PW{p, Kt, Mt};
  };
  auto packddp = [&](const DDP& d) -> DDPW {
    DDPW w;
    w.We1  = packw(d.We1,  C + 2, C);
    w.We2  = packw(d.We2,  C,     2 * C);
    w.Wrel = packw(d.Wrel, 2 * C, C);
    w.Wroot= packw(d.Wroot,2 * C, C);
    w.Wskip= packw(d.Wskip,C + 2, C);
    return w;
  };
  DDPW wbn1 = packddp(bn1),  wd0 = packddp(down0), wd1 = packddp(down1);
  DDPW wd2  = packddp(down2), wo0 = packddp(op0),   wo1 = packddp(op1);
  PW wliftW1 = packw(liftW1, 3, 256);
  PW wliftW2 = packw(liftW2, 256, C);
  PW wprojW1 = packw(projW1, C, 256);
  PW wprojW2 = packw(projW2, 256, 1);
  PW wup0W1 = packw(up0.W1, 2 * C, C), wup0W2 = packw(up0.W2, C, C);
  PW wup1W1 = packw(up1.W1, 2 * C, C), wup1W2 = packw(up1.W2, C, C);

  // gemm: N % 128 == 0; A row stride == W.Kt*32 (zero padded)
  auto gemm = [&](const float* A, int N, const PW& W, const float* bias,
                  float* Cm, int M, int act, int accum) {
    dim3 g(N / 128, W.Mt);
    k_gemm<<<g, 256, 0, stream>>>(A, W.w, W.Kt, bias, Cm, M, act, accum);
  };

  // ---------------- cluster structure CL0 (pos, 8x8, m=256) ----------------
  k_compute_cl<<<ew(N0), 256, 0, stream>>>(pos, N0, NPG0, 8, 8, cl0);
  k_fill_int<<<ew(512), 256, 0, stream>>>(cnt, 512, 0);
  k_bin<<<ew(N0), 256, 0, stream>>>(cl0, N0, 256, cnt, mem0);
  k_sort_members<<<512, 256, 0, stream>>>(mem0, 256);

  // ---------------- lift (chunked, 3 -> 256 -> 64) ----------------
  k_pad_rows<<<ew((long long)N0 * 32), 256, 0, stream>>>(x, 3, 32, bufC, N0);
  for (int c = 0; c < Bg; ++c) {
    gemm(bufC + (size_t)c * CHUNK * 32, CHUNK, wliftW1, liftb1, bufA, 256, ACT_GELU, 0);
    gemm(bufA, CHUNK, wliftW2, liftb2, bufD + (size_t)c * CHUNK * C, C, ACT_GELU, 0);
  }

  // ---------------- down0 ddop (m=256) ----------------
  k_concat2<<<ew((long long)N0 * 96), 256, 0, stream>>>(bufD, C, pos, 2, bufC, N0, 96);
  gemm(bufC, N0, wd0.We1, down0.be1, bufE, C, ACT_GELU, 0);
  gemm(bufE, N0, wd0.We2, down0.be2, bufA, 2 * C, ACT_NONE, 0);
  k_agg256<<<512, 256, 0, stream>>>(bufA, pos, mem0, bufB);
  gemm(bufC, N0, wd0.Wskip, nullptr, bufE, C, ACT_NONE, 0);
  gemm(bufB, N0, wd0.Wrel, down0.brel, bufE, C, ACT_NONE, 1);
  gemm(bufA, N0, wd0.Wroot, nullptr, bufE, C, ACT_NONE, 1);
  k_inorm<<<Bg * C, 256, 0, stream>>>(bufE, x0, NPG0, C, ACT_GELU);

  // ---------------- pool0 -> level1 ----------------
  k_pool_mean<<<512, 64, 0, stream>>>(x0, C, mem0, 256, x1p);
  k_pool_mean<<<512, 64, 0, stream>>>(pos, 2, mem0, 256, pos1);

  // ---------------- down1 ddop (m=1 -> agg==0) ----------------
  k_concat2<<<ew((long long)N1 * 96), 256, 0, stream>>>(x1p, C, pos1, 2, sInp, N1, 96);
  gemm(sInp, N1, wd1.We1, down1.be1, sT, C, ACT_GELU, 0);
  gemm(sT, N1, wd1.We2, down1.be2, sHd, 2 * C, ACT_NONE, 0);
  gemm(sInp, N1, wd1.Wskip, down1.brel, sOut, C, ACT_NONE, 0);
  gemm(sHd, N1, wd1.Wroot, nullptr, sOut, C, ACT_NONE, 1);
  k_inorm<<<Bg * C, 256, 0, stream>>>(sOut, x1, 64, C, ACT_GELU);

  // ---------------- CL1 (pos1, 4x4, m=4) + pool1 ----------------
  k_compute_cl<<<ew(N1), 256, 0, stream>>>(pos1, N1, 64, 4, 4, cl1);
  k_fill_int<<<ew(512), 256, 0, stream>>>(cnt, 512, 0);
  k_bin<<<ew(N1), 256, 0, stream>>>(cl1, N1, 4, cnt, mem1);
  k_sort_members<<<128, 256, 0, stream>>>(mem1, 4);
  k_pool_mean<<<128, 64, 0, stream>>>(x1, C, mem1, 4, x2p);
  k_pool_mean<<<128, 64, 0, stream>>>(pos1, 2, mem1, 4, pos2);

  // ---------------- down2 ddop (m=1) ----------------
  k_concat2<<<ew((long long)N2 * 96), 256, 0, stream>>>(x2p, C, pos2, 2, sInp, N2, 96);
  gemm(sInp, N2, wd2.We1, down2.be1, sT, C, ACT_GELU, 0);
  gemm(sT, N2, wd2.We2, down2.be2, sHd, 2 * C, ACT_NONE, 0);
  gemm(sInp, N2, wd2.Wskip, down2.brel, sOut, C, ACT_NONE, 0);
  gemm(sHd, N2, wd2.Wroot, nullptr, sOut, C, ACT_NONE, 1);
  k_inorm<<<Bg * C, 256, 0, stream>>>(sOut, x2, 16, C, ACT_GELU);

  // ---------------- CL2 (pos2, 1x1, m=16) ----------------
  k_compute_cl<<<ew(N2), 256, 0, stream>>>(pos2, N2, 16, 1, 1, cl2);
  k_fill_int<<<ew(512), 256, 0, stream>>>(cnt, 512, 0);
  k_bin<<<ew(N2), 256, 0, stream>>>(cl2, N2, 16, cnt, mem2);
  k_sort_members<<<8, 256, 0, stream>>>(mem2, 16);

  // ---------------- bottleneck bn1 (m=16) + residual ----------------
  k_concat2<<<ew((long long)N2 * 96), 256, 0, stream>>>(x2, C, pos2, 2, sInp, N2, 96);
  gemm(sInp, N2, wbn1.We1, bn1.be1, sT, C, ACT_GELU, 0);
  gemm(sT, N2, wbn1.We2, bn1.be2, sHd, 2 * C, ACT_NONE, 0);
  k_agg_small<<<ew(N2 * 128), 256, 0, stream>>>(sHd, pos2, mem2, cl2, sAgg, N2, 16);
  gemm(sInp, N2, wbn1.Wskip, nullptr, sOut, C, ACT_NONE, 0);
  gemm(sAgg, N2, wbn1.Wrel, bn1.brel, sOut, C, ACT_NONE, 1);
  gemm(sHd, N2, wbn1.Wroot, nullptr, sOut, C, ACT_NONE, 1);
  k_inorm<<<Bg * C, 256, 0, stream>>>(sOut, hb, 16, C, ACT_GELU);
  k_add<<<ew(N2 * C), 256, 0, stream>>>(x2, hb, x2, N2 * C);

  // ---------------- up0: knn(level2 -> level1), MLP, op0 ----------------
  k_knn3_find<<<ew(N1), 256, 0, stream>>>(pos2, pos1, 16, 64, N1, knni, knnw);
  k_knn3_apply<<<ew(N1 * C), 256, 0, stream>>>(x2, knni, knnw, xi0, N1, C);
  k_concat2<<<ew(N1 * 2 * C), 256, 0, stream>>>(xi0, C, x1, C, cat0, N1, 2 * C);
  gemm(cat0, N1, wup0W1, up0.b1, sT, C, ACT_NONE, 0);
  k_colstats<<<C, 256, 0, stream>>>(sT, N1, C, stats, stats + 64);
  k_bn_relu<<<ew(N1 * C), 256, 0, stream>>>(sT, xm0, N1, C, stats, stats + 64, up0.g, up0.bt);
  gemm(xm0, N1, wup0W2, up0.b2, sOut, C, ACT_NONE, 0);
  k_inorm<<<Bg * C, 256, 0, stream>>>(sOut, xm0, 64, C, ACT_GELU);
  k_add<<<ew(N1 * C), 256, 0, stream>>>(xm0, x1, opin0, N1 * C);
  // op0 ddop (m=4)
  k_concat2<<<ew((long long)N1 * 96), 256, 0, stream>>>(opin0, C, pos1, 2, sInp, N1, 96);
  gemm(sInp, N1, wo0.We1, op0.be1, sT, C, ACT_GELU, 0);
  gemm(sT, N1, wo0.We2, op0.be2, sHd, 2 * C, ACT_NONE, 0);
  k_agg_small<<<ew(N1 * 128), 256, 0, stream>>>(sHd, pos1, mem1, cl1, sAgg, N1, 4);
  gemm(sInp, N1, wo0.Wskip, nullptr, sOut, C, ACT_NONE, 0);
  gemm(sAgg, N1, wo0.Wrel, op0.brel, sOut, C, ACT_NONE, 1);
  gemm(sHd, N1, wo0.Wroot, nullptr, sOut, C, ACT_NONE, 1);
  k_inorm<<<Bg * C, 256, 0, stream>>>(sOut, xo1, 64, C, ACT_GELU);

  // ---------------- up1: knn(level1 -> level0), MLP, op1 ----------------
  k_knn3_find<<<ew(N0), 256, 0, stream>>>(pos1, pos, 64, NPG0, N0, knni, knnw);
  k_knn3_apply<<<ew((long long)N0 * C), 256, 0, stream>>>(xo1, knni, knnw, bufD, N0, C);
  k_concat2<<<ew((long long)N0 * 2 * C), 256, 0, stream>>>(bufD, C, x0, C, bufA, N0, 2 * C);
  gemm(bufA, N0, wup1W1, up1.b1, bufE, C, ACT_NONE, 0);
  k_colstats<<<C, 256, 0, stream>>>(bufE, N0, C, stats, stats + 64);
  k_bn_relu<<<ew((long long)N0 * C), 256, 0, stream>>>(bufE, bufD, N0, C, stats, stats + 64, up1.g, up1.bt);
  gemm(bufD, N0, wup1W2, up1.b2, bufE, C, ACT_NONE, 0);
  k_inorm<<<Bg * C, 256, 0, stream>>>(bufE, bufD, NPG0, C, ACT_GELU);
  k_add<<<ew((long long)N0 * C), 256, 0, stream>>>(bufD, x0, bufE, N0 * C);
  // op1 ddop (m=256), no norm/act
  k_concat2<<<ew((long long)N0 * 96), 256, 0, stream>>>(bufE, C, pos, 2, bufC, N0, 96);
  gemm(bufC, N0, wo1.We1, op1.be1, bufD, C, ACT_GELU, 0);
  gemm(bufD, N0, wo1.We2, op1.be2, bufA, 2 * C, ACT_NONE, 0);
  k_agg256<<<512, 256, 0, stream>>>(bufA, pos, mem0, bufB);
  gemm(bufC, N0, wo1.Wskip, nullptr, bufE, C, ACT_NONE, 0);
  gemm(bufB, N0, wo1.Wrel, op1.brel, bufE, C, ACT_NONE, 1);
  gemm(bufA, N0, wo1.Wroot, nullptr, bufE, C, ACT_NONE, 1);

  // ---------------- proj (chunked, 64 -> 256 -> 1) ----------------
  float* outp = (float*)d_out;
  for (int c = 0; c < Bg; ++c) {
    gemm(bufE + (size_t)c * CHUNK * C, CHUNK, wprojW1, projb1, bufD, 256, ACT_GELU, 0);
    gemm(bufD, CHUNK, wprojW2, projb2, outp + (size_t)c * CHUNK, 1, ACT_NONE, 0);
  }
}
